// DynamicsFunction_42253888258333
// MI455X (gfx1250) — matrix operand order repaired and checked by reading, not compiled, not run on hardware
//
#include <hip/hip_runtime.h>
#include <hip/hip_bf16.h>
#include <math.h>

typedef _Float16 h16;
typedef __attribute__((ext_vector_type(16))) _Float16 v16h;
typedef __attribute__((ext_vector_type(8)))  _Float16 v8h;
typedef __attribute__((ext_vector_type(8)))  float    v8f;

#define BATCH_M 8192
#define HID     512
#define SDIM    256
#define N_STEPS 16

#define MTILE 128
#define NTILE 64
#define KSTEP 32
#define LDS_LDA 40   // halfs per A row in LDS (80B, conflict-free for b128)
#define LDS_LDB 40   // halfs per B row in LDS

enum { MODE_TANH = 0, MODE_INIT = 1, MODE_RK4 = 2, MODE_OUT = 3 };

__device__ __forceinline__ v8f wmma_f32_16x16x32(v16h a, v16h b, v8f c) {
  // 8 args: (neg_a, A, neg_b, B, c_mod, C, reuse_a, reuse_b)
  return __builtin_amdgcn_wmma_f32_16x16x32_f16(false, a, false, b, (short)0, c,
                                                false, false);
}

// Hardware-transcendental tanh: tanh(x) = 1 - 2/(exp2(2*log2e*x)+1).
// v_exp_f32 + v_rcp_f32 on the TRANS pipe (co-executes with WMMA per ISA),
// ~5 ops vs ~25 for OCML tanhf; saturates correctly to +/-1.
__device__ __forceinline__ float fast_tanh(float x) {
  const float two_log2e = 2.8853900817779268f;  // 2*log2(e)
  float e = __builtin_amdgcn_exp2f(x * two_log2e);
  return 1.0f - 2.0f * __builtin_amdgcn_rcpf(e + 1.0f);
}

union FragH {
  v16h v;
  struct { v8h lo; v8h hi; } p;
};

// C = epilogue(A[M,K] @ Wt^T + bias), Wt stored as [N][K] f16.
// Wave32 WMMA 16x16x32 f16 -> f32. 128x64 tile per 256-thread block,
// 8 waves in a 4(M) x 2(N) grid, each wave owns 2x2 16x16 accumulators.
template <int MODE>
__global__ __launch_bounds__(256, 2)
void gemm_rk4(const h16* __restrict__ A, const h16* __restrict__ Wt,
              const float* __restrict__ bias,
              h16* __restrict__ outH, float* __restrict__ Y,
              float* __restrict__ ACC, float* __restrict__ outF,
              const int* __restrict__ evolve_ptr,
              int K, int N, float cmul, float alpha, int acc_load, int finalize) {
  __shared__ __align__(32) h16 As[2][MTILE * LDS_LDA];
  __shared__ __align__(32) h16 Bs[2][NTILE * LDS_LDB];

  const int tid = threadIdx.x;
  const int mblocks = BATCH_M / MTILE;
  const int mblock = blockIdx.x % mblocks;
  const int nblock = blockIdx.x / mblocks;
  const int mbase = mblock * MTILE;
  const int nbase = nblock * NTILE;

  const int lane  = tid & 31;
  const int wid   = tid >> 5;
  const int waveM = (wid & 3) * 32;   // 4 waves cover 128 rows
  const int waveN = (wid >> 2) * 32;  // 2 waves cover 64 cols
  const int lrow  = lane & 15;
  const int lhalf = lane >> 4;

  v8f c[2][2] = {};

  auto loadA = [&](int buf, int kb) {
    // 128 rows x 32 halfs = 512 8-half chunks; 2 per thread
#pragma unroll
    for (int i = 0; i < 2; ++i) {
      int idx = tid + i * 256;
      int row = idx >> 2;
      int ck  = (idx & 3) * 8;
      *(v8h*)&As[buf][row * LDS_LDA + ck] =
          *(const v8h*)(A + (size_t)(mbase + row) * K + kb + ck);
    }
  };
  auto loadB = [&](int buf, int kb) {
    // 64 rows x 32 halfs = 256 chunks; 1 per thread
    int row = tid >> 2;
    int ck  = (tid & 3) * 8;
    *(v8h*)&Bs[buf][row * LDS_LDB + ck] =
        *(const v8h*)(Wt + (size_t)(nbase + row) * K + kb + ck);
  };

  loadA(0, 0);
  loadB(0, 0);
  __syncthreads();

  const int nk = K / KSTEP;
  for (int kt = 0; kt < nk; ++kt) {
    const int buf = kt & 1;
    if (kt + 1 < nk) {  // prefetch next K-slab into the other buffer
      loadA(buf ^ 1, (kt + 1) * KSTEP);
      loadB(buf ^ 1, (kt + 1) * KSTEP);
    }

    FragH a[2], b[2];
#pragma unroll
    for (int mi = 0; mi < 2; ++mi) {
      // A 16x32 f16 layout: lanes 0-15 hold K{0..7,16..23}, lanes 16-31 K{8..15,24..31}
      const h16* p = &As[buf][(waveM + mi * 16 + lrow) * LDS_LDA];
      a[mi].p.lo = *(const v8h*)(p + 8 * lhalf);
      a[mi].p.hi = *(const v8h*)(p + 16 + 8 * lhalf);
    }
#pragma unroll
    for (int ni = 0; ni < 2; ++ni) {
      // B 32x16 f16: lane holds column lrow, lanes 0-15 K{0..7,16..23}, lanes 16-31 K{8..15,24..31}
      const h16* p = &Bs[buf][(waveN + ni * 16 + lrow) * LDS_LDB];
      b[ni].p.lo = *(const v8h*)(p + 8 * lhalf);
      b[ni].p.hi = *(const v8h*)(p + 16 + 8 * lhalf);
    }
#pragma unroll
    for (int mi = 0; mi < 2; ++mi)
#pragma unroll
      for (int ni = 0; ni < 2; ++ni)
        c[mi][ni] = wmma_f32_16x16x32(a[mi].v, b[ni].v, c[mi][ni]);

    __syncthreads();
  }

  float dt = 0.0f;
  if (MODE == MODE_RK4)
    dt = (float)(*evolve_ptr) * (1.0f / (float)N_STEPS);

  // C/D layout: lane l -> col = l&15; rows (l>>4)*8 + r for VGPR r
#pragma unroll
  for (int mi = 0; mi < 2; ++mi) {
#pragma unroll
    for (int ni = 0; ni < 2; ++ni) {
      const int col = nbase + waveN + ni * 16 + lrow;
      const int rb  = mbase + waveM + mi * 16 + lhalf * 8;
      const float bv = bias[col];
#pragma unroll
      for (int r = 0; r < 8; ++r) {
        const int row = rb + r;
        const size_t idx = (size_t)row * N + col;
        float v = c[mi][ni][r] + bv;
        if (MODE == MODE_TANH) {
          outH[idx] = (h16)fast_tanh(v);
        } else if (MODE == MODE_INIT) {
          Y[idx] = v;
          outH[idx] = (h16)v;
        } else if (MODE == MODE_OUT) {
          outF[idx] = v;
        } else {  // MODE_RK4: v == k_e[row,col]
          const float k = v;
          float a0 = acc_load ? ACC[idx] : 0.0f;
          float an = a0 + alpha * k;
          if (!finalize) {
            ACC[idx] = an;
            outH[idx] = (h16)(Y[idx] + cmul * dt * k);  // next field input
          } else {
            float yn = Y[idx] + (dt * (1.0f / 6.0f)) * an;
            Y[idx] = yn;
            outH[idx] = (h16)yn;  // becomes next step's eval-1 input
          }
        }
      }
    }
  }
}

__global__ void cvt_f16(const float* __restrict__ in, h16* __restrict__ out, int n) {
  int i = blockIdx.x * 256 + threadIdx.x;
  if (i < n) out[i] = (h16)in[i];
}

// W[K][N] f32 -> Wt[N][K] f16 (so B columns are contiguous in K)
__global__ void transpose_cvt(const float* __restrict__ W, h16* __restrict__ Wt,
                              int K, int N) {
  int i = blockIdx.x * 256 + threadIdx.x;
  if (i < K * N) {
    int k = i / N, n = i % N;
    Wt[(size_t)n * K + k] = (h16)W[i];
  }
}

extern "C" void kernel_launch(void* const* d_in, const int* in_sizes, int n_in,
                              void* d_out, int out_size, void* d_ws, size_t ws_size,
                              hipStream_t stream) {
  (void)in_sizes; (void)n_in; (void)out_size; (void)ws_size;
  const float* x    = (const float*)d_in[0];
  const float* W1   = (const float*)d_in[1];
  const float* b1   = (const float*)d_in[2];
  const float* W2   = (const float*)d_in[3];
  const float* b2   = (const float*)d_in[4];
  const float* We1  = (const float*)d_in[5];
  const float* be1  = (const float*)d_in[6];
  const float* We2  = (const float*)d_in[7];
  const float* be2  = (const float*)d_in[8];
  const float* Wr   = (const float*)d_in[9];
  const float* br   = (const float*)d_in[10];
  const int* evolve = (const int*)d_in[11];

  char* ws = (char*)d_ws;
  size_t off = 0;
  auto alloc = [&](size_t bytes) -> char* {
    char* p = ws + off;
    off += (bytes + 255) & ~(size_t)255;
    return p;
  };

  h16* xh    = (h16*)alloc((size_t)BATCH_M * SDIM * 2);
  h16* W1t   = (h16*)alloc((size_t)HID * SDIM * 2);
  h16* W2t   = (h16*)alloc((size_t)HID * HID * 2);
  h16* We1t  = (h16*)alloc((size_t)HID * HID * 2);
  h16* We2t  = (h16*)alloc((size_t)HID * HID * 2);
  h16* Wrt   = (h16*)alloc((size_t)SDIM * HID * 2);
  h16* T     = (h16*)alloc((size_t)BATCH_M * HID * 2);  // tanh hidden, f16
  h16* INh   = (h16*)alloc((size_t)BATCH_M * HID * 2);  // f16 field input / f16(y)
  float* Y   = (float*)alloc((size_t)BATCH_M * HID * 4);  // f32 RK4 state
  float* ACC = (float*)alloc((size_t)BATCH_M * HID * 4);  // f32 k accumulator

  dim3 blk(256);
  cvt_f16<<<(BATCH_M * SDIM + 255) / 256, blk, 0, stream>>>(x, xh, BATCH_M * SDIM);
  transpose_cvt<<<(SDIM * HID + 255) / 256, blk, 0, stream>>>(W1, W1t, SDIM, HID);
  transpose_cvt<<<(HID * HID + 255) / 256, blk, 0, stream>>>(W2, W2t, HID, HID);
  transpose_cvt<<<(HID * HID + 255) / 256, blk, 0, stream>>>(We1, We1t, HID, HID);
  transpose_cvt<<<(HID * HID + 255) / 256, blk, 0, stream>>>(We2, We2t, HID, HID);
  transpose_cvt<<<(HID * SDIM + 255) / 256, blk, 0, stream>>>(Wr, Wrt, HID, SDIM);

  const int mblocks = BATCH_M / MTILE;
  auto grid_for = [&](int N) { return dim3(mblocks * (N / NTILE)); };

  // Embedding: tanh(x@W1+b1)@W2+b2 -> y0 (f32) and f16 copy
  gemm_rk4<MODE_TANH><<<grid_for(HID), blk, 0, stream>>>(
      xh, W1t, b1, T, nullptr, nullptr, nullptr, nullptr, SDIM, HID, 0, 0, 0, 0);
  gemm_rk4<MODE_INIT><<<grid_for(HID), blk, 0, stream>>>(
      T, W2t, b2, INh, Y, nullptr, nullptr, nullptr, HID, HID, 0, 0, 0, 0);

  // RK4: 16 steps x 4 field evals, combine fused into GEMM2 epilogue
  const float cmul[4]  = {0.5f, 0.5f, 1.0f, 0.0f};
  const float alpha[4] = {1.0f, 2.0f, 2.0f, 1.0f};
  for (int s = 0; s < N_STEPS; ++s) {
    for (int e = 0; e < 4; ++e) {
      gemm_rk4<MODE_TANH><<<grid_for(HID), blk, 0, stream>>>(
          INh, We1t, be1, T, nullptr, nullptr, nullptr, nullptr,
          HID, HID, 0, 0, 0, 0);
      gemm_rk4<MODE_RK4><<<grid_for(HID), blk, 0, stream>>>(
          T, We2t, be2, INh, Y, ACC, nullptr, evolve,
          HID, HID, cmul[e], alpha[e], e > 0 ? 1 : 0, e == 3 ? 1 : 0);
    }
  }

  // Readout: yT @ Wr + br -> f32 d_out [8192,256]
  gemm_rk4<MODE_OUT><<<grid_for(SDIM), blk, 0, stream>>>(
      INh, Wrt, br, nullptr, nullptr, nullptr, (float*)d_out, nullptr,
      HID, SDIM, 0, 0, 0, 0);
}
